// Attention_15410342658774
// MI455X (gfx1250) — compile-verified, matrix operand order checked by reading, not run on hardware
//
#include <hip/hip_runtime.h>
#include <hip/hip_bf16.h>
#include <math.h>

#define B_    64
#define T_    1000
#define E_    512
#define EL_   512
#define RNN_  1024
#define AD_   128
#define NF_   32
#define K_    31
#define PAD_  15
#define TT_   16
#define NTT_  ((T_ + TT_ - 1) / TT_)   // 63

// TDM LDS padding: pad_interval=7 (256 DWORDs), pad_amount=3 (4 DWORDs)
// -> chunk stride 260 DWORDs (16B aligned), row stride 520 DWORDs
#define CHUNK_ 260
#define ROWSTR_ 520

typedef __attribute__((ext_vector_type(8)))  float        v8f;
typedef __attribute__((ext_vector_type(4)))  float        f4;
typedef __attribute__((ext_vector_type(16))) __bf16       v16bf;
typedef __attribute__((ext_vector_type(4)))  unsigned int u32x4;
typedef __attribute__((ext_vector_type(8)))  int          i32x8;

// -------------------------------------------------------------------------
// Kernel 0: convert projection weights to bf16 once (they are reused by
// every workgroup; bf16 rows let the WMMA B-fragments load without cvt).
// -------------------------------------------------------------------------
__global__ void prep_w_kernel(const float* __restrict__ W1,
                              const float* __restrict__ W2,
                              __bf16* __restrict__ o1,
                              __bf16* __restrict__ o2) {
    int i = blockIdx.x * blockDim.x + threadIdx.x;
    if (i < AD_ * E_) {
        o1[i] = (__bf16)W1[i];
        o2[i] = (__bf16)W2[i];
    }
}

// -------------------------------------------------------------------------
// Kernel 1: pq = query @ query_W^T and spq = query @ self_query_W^T  (B,AD)
// -------------------------------------------------------------------------
__global__ void pq_kernel(const float* __restrict__ query,
                          const float* __restrict__ qW,
                          const float* __restrict__ sqW,
                          float* __restrict__ pq,
                          float* __restrict__ spq) {
    int idx = blockIdx.x * blockDim.x + threadIdx.x;
    if (idx >= 2 * B_ * AD_) return;
    int sel = idx / (B_ * AD_);
    int r   = idx % (B_ * AD_);
    int b = r / AD_, a = r % AD_;
    const float* W = sel ? sqW : qW;
    const float* q = query + (size_t)b * RNN_;
    const float* w = W + (size_t)a * RNN_;
    float acc = 0.f;
    for (int k = 0; k < RNN_; k += 4) {
        f4 qv = *(const f4*)(q + k);
        f4 wv = *(const f4*)(w + k);
        acc += qv.x * wv.x + qv.y * wv.y + qv.z * wv.z + qv.w * wv.w;
    }
    (sel ? spq : pq)[r] = acc;
}

// -------------------------------------------------------------------------
// Kernel 2: energies[b,t] = sum_a v[a] * tanh(pq[b,a] + proc[b,t,a] + pm[b,t,a])
// A-tile (16 x F f32) streamed into LDS by the Tensor Data Mover (one
// tensor_load_to_lds per workgroup, TENSORcnt-tracked, OOB rows zero-filled,
// bank-conflict padding done by the TDM). pm via bf16 WMMA, f32 accumulate.
// grid = (NTT_, B_), block = 256 (8 waves; wave w owns AD columns [16w,16w+16))
// -------------------------------------------------------------------------
template <bool HAS_PROC>
__global__ void __launch_bounds__(256)
energies_kernel(const float* __restrict__ mem,    // (B,T,F) f32
                const __bf16* __restrict__ Wbf,   // (AD,F) bf16
                const float* __restrict__ pq,     // (B,AD)
                const float* __restrict__ v,      // (AD)
                const float* __restrict__ aw,     // (B,T)
                const float* __restrict__ awcum,  // (B,T)
                const float* __restrict__ convW,  // (NF,2,K)
                const float* __restrict__ denseW, // (AD,NF)
                float* __restrict__ energies,     // (B,T)
                int F) {
    __shared__ float AS[TT_ * ROWSTR_];           // TDM-padded A tile (33.3 KB)
    __shared__ float convS[NF_][TT_];
    __shared__ float denseS[AD_ * NF_];
    __shared__ float vS[AD_];
    __shared__ float pqS[AD_];
    __shared__ float eacc[TT_];

    const int b    = blockIdx.y;
    const int t0   = blockIdx.x * TT_;
    const int tid  = threadIdx.x;
    const int wave = tid >> 5;
    const int lane = tid & 31;
    const int half = lane >> 4;
    const int l15  = lane & 15;

    if (tid < AD_) { vS[tid] = v[tid]; pqS[tid] = pq[b * AD_ + tid]; }
    if (tid < TT_) eacc[tid] = 0.f;

    // ---- TDM: DMA the 16 x F f32 activation tile into LDS (wave 0 issues) --
    if (wave == 0) {
        unsigned lds0 = (unsigned)(unsigned long long)(const void*)&AS[0];
        unsigned long long ga =
            (unsigned long long)(const void*)(mem + ((size_t)b * T_ + t0) * F);
        unsigned rows = (unsigned)(T_ - t0);      // OOB rows -> zero fill

        u32x4 g0;
        g0[0] = 1u;                                           // count=1
        g0[1] = lds0;                                         // lds_addr
        g0[2] = (unsigned)(ga & 0xFFFFFFFFull);               // global_addr lo
        g0[3] = (unsigned)((ga >> 32) & 0x1FFFFFFull)         // global_addr hi
              | (2u << 30);                                   // type=2 (image)

        i32x8 g1;
        g1[0] = (int)((2u << 16)      // data_size = 4B
                    | (1u << 20)      // pad_enable
                    | (7u << 22)      // pad_interval: 256 DWORDs
                    | (3u << 25));    // pad_amount: 4 DWORDs
        g1[1] = (int)((unsigned)F << 16);     // tensor_dim0 (bits 63:48)
        g1[2] = (int)(rows << 16);            // tensor_dim1 (bits 95:80)
        g1[3] = (int)((unsigned)F << 16);     // tile_dim0   (bits 127:112)
        g1[4] = (int)TT_;                     // tile_dim1 = 16
        g1[5] = (int)F;                       // tensor_dim0_stride
        g1[6] = 0;
        g1[7] = 0;

        asm volatile("tensor_load_to_lds %0, %1"
                     :: "s"(g0), "s"(g1) : "memory");
        __builtin_amdgcn_s_wait_tensorcnt(0);
    }

    if (HAS_PROC) {
        for (int i = tid; i < AD_ * NF_; i += 256) denseS[i] = denseW[i];
        for (int i = tid; i < NF_ * TT_; i += 256) {
            int f = i / TT_, tl = i % TT_;
            int t = t0 + tl;
            const float* w0 = convW + f * 2 * K_;
            float acc = 0.f;
            for (int kk = 0; kk < K_; ++kk) {
                int ts = t + kk - PAD_;
                if (ts >= 0 && ts < T_) {
                    acc += w0[kk]      * aw[b * T_ + ts]
                         + w0[K_ + kk] * awcum[b * T_ + ts];
                }
            }
            convS[f][tl] = acc;
        }
    }
    __syncthreads();   // TDM data + conv staging visible to all waves

    const int acol = wave * 16 + l15;                 // B-matrix column (N)
    const __bf16* Brow = Wbf + (size_t)acol * F;
    const int koff = half * 8;                        // ISA bf16 lane K offset

    v8f c = {0.f, 0.f, 0.f, 0.f, 0.f, 0.f, 0.f, 0.f};

#pragma unroll 4
    for (int kb = 0; kb < F; kb += 32) {
        const int col = kb + koff;
        // padded-LDS index: never crosses a 256-DWORD chunk (col base % 8 == 0)
        const int idx = l15 * ROWSTR_ + (col >> 8) * CHUNK_ + (col & 255);
        f4 a0 = *(const f4*)(AS + idx);
        f4 a1 = *(const f4*)(AS + idx + 4);
        f4 a2 = *(const f4*)(AS + idx + 16);
        f4 a3 = *(const f4*)(AS + idx + 20);

        union { v16bf v; f4 q[2]; } bu;
        bu.q[0] = *(const f4*)(Brow + col);           // 8 bf16 = 16B
        bu.q[1] = *(const f4*)(Brow + col + 16);

        v16bf af;
#pragma unroll
        for (int i = 0; i < 4; ++i) {
            af[i]      = (__bf16)a0[i];
            af[4 + i]  = (__bf16)a1[i];
            af[8 + i]  = (__bf16)a2[i];
            af[12 + i] = (__bf16)a3[i];
        }
        c = __builtin_amdgcn_wmma_f32_16x16x32_bf16(
                false, af, false, bu.v, (short)0, c, false, false);
    }

    // epilogue: c[r] = pm[t0 + r + 8*half][acol]
#pragma unroll
    for (int r = 0; r < 8; ++r) {
        const int M = r + half * 8;
        float val = c[r] + pqS[acol];
        if (HAS_PROC) {
            float pv = 0.f;
#pragma unroll
            for (int f = 0; f < NF_; ++f) pv += denseS[acol * NF_ + f] * convS[f][M];
            val += pv;
        }
        float e = tanhf(val) * vS[acol];
        e += __shfl_xor(e, 1);
        e += __shfl_xor(e, 2);
        e += __shfl_xor(e, 4);
        e += __shfl_xor(e, 8);
        if (l15 == 0) atomicAdd(&eacc[M], e);
    }
    __syncthreads();
    if (tid < TT_ && (t0 + tid) < T_) energies[b * T_ + t0 + tid] = eacc[tid];
}

// -------------------------------------------------------------------------
__device__ __forceinline__ float blockReduceSum(float v, float* red) {
#pragma unroll
    for (int m = 1; m < 32; m <<= 1) v += __shfl_xor(v, m);
    __syncthreads();
    if ((threadIdx.x & 31) == 0) red[threadIdx.x >> 5] = v;
    __syncthreads();
    float s = red[0];
#pragma unroll
    for (int i = 1; i < 8; ++i) s += red[i];
    return s;
}

// -------------------------------------------------------------------------
// Kernel 3: per-batch normalization, alpha recurrence, contexts, ta, packing.
// out[b] = [context(512) | alignments(1000) | u_new(1) | cum_new(1000)
//           | ctx2(512) | w2(1000)]  (stride 4025)
// -------------------------------------------------------------------------
__global__ void __launch_bounds__(256)
finalize_kernel(const float* __restrict__ energies,
                const float* __restrict__ e2,
                const float* __restrict__ mem,
                const float* __restrict__ smem,
                const float* __restrict__ awcum,
                const float* __restrict__ alpha,
                const float* __restrict__ uu,
                const float* __restrict__ query,
                const float* __restrict__ taW,
                const float* __restrict__ tab,
                float* __restrict__ out) {
    __shared__ float wS[T_];
    __shared__ float red[8];

    const int b   = blockIdx.x;
    const int tid = threadIdx.x;
    float* obase = out + (size_t)b * (E_ + T_ + 1 + T_ + EL_ + T_);

    float ssum = 0.f;
    for (int t = tid; t < T_; t += 256) {
        float s = 1.f / (1.f + __expf(-energies[b * T_ + t]));
        wS[t] = s;
        ssum += s;
    }
    ssum = blockReduceSum(ssum, red);

    const float u = uu[b];
    float asum = 0.f;
    for (int t = tid; t < T_; t += 256) {
        float align = wS[t] / ssum;
        obase[E_ + T_ + 1 + t] = awcum[b * T_ + t] + align;           // cum_new
        float ash = (t == 0) ? 0.f : alpha[b * T_ + t - 1];
        float an  = ((1.f - u) * alpha[b * T_ + t] + u * ash + 1e-8f) * align;
        wS[t] = an;
        asum += an;
    }
    asum = blockReduceSum(asum, red);
    for (int t = tid; t < T_; t += 256) {
        float al = wS[t] / asum;
        wS[t] = al;
        obase[E_ + t] = al;                                           // alignments
    }
    __syncthreads();

    float c0 = 0.f, c1 = 0.f;
    for (int t = 0; t < T_; ++t) {
        const float* row = mem + ((size_t)b * T_ + t) * E_;
        __builtin_prefetch(row + E_, 0, 0);
        float w = wS[t];
        c0 += w * row[tid];
        c1 += w * row[tid + 256];
    }
    obase[tid]       = c0;                                            // context
    obase[tid + 256] = c1;

    float part = taW[tid] * c0 + taW[tid + 256] * c1;
    for (int j = tid; j < RNN_; j += 256) part += taW[E_ + j] * query[b * RNN_ + j];
    part = blockReduceSum(part, red);
    if (tid == 0) obase[E_ + T_] = 1.f / (1.f + __expf(-(part + tab[0])));

    float s2 = 0.f;
    for (int t = tid; t < T_; t += 256) {
        float s = 1.f / (1.f + __expf(-e2[b * T_ + t]));
        wS[t] = s;
        s2 += s;
    }
    s2 = blockReduceSum(s2, red);
    for (int t = tid; t < T_; t += 256) {
        float w = wS[t] / s2;
        wS[t] = w;
        obase[E_ + T_ + 1 + T_ + EL_ + t] = w;                        // w2
    }
    __syncthreads();

    float d0 = 0.f, d1 = 0.f;
    for (int t = 0; t < T_; ++t) {
        const float* row = smem + ((size_t)b * T_ + t) * EL_;
        __builtin_prefetch(row + EL_, 0, 0);
        float w = wS[t];
        d0 += w * row[tid];
        d1 += w * row[tid + 256];
    }
    obase[E_ + T_ + 1 + T_ + tid]       = d0;                         // ctx2
    obase[E_ + T_ + 1 + T_ + tid + 256] = d1;
}

// -------------------------------------------------------------------------
extern "C" void kernel_launch(void* const* d_in, const int* in_sizes, int n_in,
                              void* d_out, int out_size, void* d_ws, size_t ws_size,
                              hipStream_t stream) {
    const float* query         = (const float*)d_in[0];
    const float* memory        = (const float*)d_in[1];
    const float* self_memory   = (const float*)d_in[2];
    const float* aw            = (const float*)d_in[3];
    const float* awcum         = (const float*)d_in[4];
    const float* alpha         = (const float*)d_in[5];
    const float* u             = (const float*)d_in[6];
    const float* memory_W      = (const float*)d_in[7];
    const float* query_W       = (const float*)d_in[8];
    const float* v_W           = (const float*)d_in[9];
    const float* loc_conv_W    = (const float*)d_in[10];
    const float* loc_dense_W   = (const float*)d_in[11];
    const float* ta_W          = (const float*)d_in[12];
    const float* ta_b          = (const float*)d_in[13];
    const float* self_memory_W = (const float*)d_in[14];
    const float* self_query_W  = (const float*)d_in[15];
    const float* self_v_W      = (const float*)d_in[16];

    float* out = (float*)d_out;
    float* ws  = (float*)d_ws;
    float* pq       = ws;                                  // B*AD
    float* spq      = ws + B_ * AD_;                       // B*AD
    float* energies = ws + 2 * B_ * AD_;                   // B*T
    float* e2       = ws + 2 * B_ * AD_ + B_ * T_;         // B*T
    __bf16* w1bf    = (__bf16*)(ws + 2 * B_ * AD_ + 2 * B_ * T_);  // AD*E
    __bf16* w2bf    = w1bf + AD_ * E_;                              // AD*EL

    prep_w_kernel<<<dim3((AD_ * E_ + 255) / 256), 256, 0, stream>>>(
        memory_W, self_memory_W, w1bf, w2bf);

    pq_kernel<<<dim3((2 * B_ * AD_ + 255) / 256), 256, 0, stream>>>(
        query, query_W, self_query_W, pq, spq);

    energies_kernel<true><<<dim3(NTT_, B_), 256, 0, stream>>>(
        memory, w1bf, pq, v_W, aw, awcum, loc_conv_W, loc_dense_W,
        energies, E_);

    energies_kernel<false><<<dim3(NTT_, B_), 256, 0, stream>>>(
        self_memory, w2bf, spq, self_v_W, nullptr, nullptr,
        nullptr, nullptr, e2, EL_);

    finalize_kernel<<<dim3(B_), 256, 0, stream>>>(
        energies, e2, memory, self_memory, awcum, alpha, u, query,
        ta_W, ta_b, out);
}
